// QueryPoolReadout_29377576305110
// MI455X (gfx1250) — compile-verified, matrix operand order checked by reading, not run on hardware
//
#include <hip/hip_runtime.h>
#include <hip/hip_bf16.h>
#include <math.h>

// Problem constants from the reference
#define N_Q   1024
#define S_KV  512
#define D_DIM 256
#define P_POOL 32
#define C_CLS 1000

typedef __attribute__((ext_vector_type(16))) __bf16 v16bf;
typedef __attribute__((ext_vector_type(2)))  __bf16 bf16x2;
typedef __attribute__((ext_vector_type(8)))  float  v8f;

union Frag {
    unsigned int u[8];
    v16bf v;
};

// ---- fp32 -> bf16 split helpers using NATIVE hardware conversions (RNE) -----
__device__ __forceinline__ unsigned int pack2_bf16(float x0, float x1) {
    bf16x2 v = { (__bf16)x0, (__bf16)x1 };
    return __builtin_bit_cast(unsigned int, v);
}

// Split two fp32 values into packed-bf16 hi and lo dwords (for B fragments).
__device__ __forceinline__ void split_pack(float x0, float x1,
                                           unsigned int& hi, unsigned int& lo) {
    __bf16 h0 = (__bf16)x0;
    __bf16 h1 = (__bf16)x1;
    bf16x2 hv = { h0, h1 };
    hi = __builtin_bit_cast(unsigned int, hv);
    lo = pack2_bf16(x0 - (float)h0, x1 - (float)h1);
}

// Scalar split for the cooperative packing phases.
__device__ __forceinline__ void split1(float x, unsigned short& hb,
                                       unsigned short& lb) {
    __bf16 h = (__bf16)x;
    hb = __builtin_bit_cast(unsigned short, h);
    __bf16 l = (__bf16)(x - (float)h);
    lb = __builtin_bit_cast(unsigned short, l);
}

// A-matrix (16x32 bf16) fragment from a PRE-PACKED bf16 (ushort) matrix in LDS.
// Fragment dwords are two contiguous 16B runs -> two ds_load_b128 per frag.
// ISA layout: lanes 0-15 row=lane (k base +0), lanes 16-31 row=lane-16 (k base +8);
// VGPR j<4: k = k0 + 8g + 2j ; VGPR j>=4: k = k0 + 16 + 8g + 2(j-4).
__device__ __forceinline__ void load_a_packed(const unsigned short* base, int row,
                                              int ld, int k0, int g, Frag& f) {
    const unsigned short* rp = base + row * ld + k0 + g * 8;
    uint4 a = *(const uint4*)rp;
    uint4 b = *(const uint4*)(rp + 16);
    f.u[0] = a.x; f.u[1] = a.y; f.u[2] = a.z; f.u[3] = a.w;
    f.u[4] = b.x; f.u[5] = b.y; f.u[6] = b.z; f.u[7] = b.w;
}

// B-matrix (32x16) fragment for GEMM1: B[k][c] = K[n][s][k0+k], row-contiguous
// per lane (column c == lane&15 selects the K row s). Converted once per element.
__device__ __forceinline__ void load_bK(const float* __restrict__ Kn, int s,
                                        int k0, int g, Frag& bh, Frag& bl) {
    const float* rp = Kn + (size_t)s * D_DIM;
    __builtin_prefetch(rp + k0 + 32, 0, 0);   // next k-chunk of this row
#pragma unroll
    for (int j = 0; j < 8; ++j) {
        int koff = ((j < 4) ? 0 : 16) + g * 8 + 2 * (j & 3);
        split_pack(rp[k0 + koff], rp[k0 + koff + 1], bh.u[j], bl.u[j]);
    }
}

// B-matrix (32x16) fragment for GEMM2: B[k][c] = V[n][k0+k][d0+c];
// column c is the d-coordinate, rows stride by D in memory.
__device__ __forceinline__ void load_bV(const float* __restrict__ Vn, int d,
                                        int k0, int g, Frag& bh, Frag& bl) {
    __builtin_prefetch(Vn + (size_t)(k0 + 32) * D_DIM + d, 0, 0);  // next s-chunk
#pragma unroll
    for (int j = 0; j < 8; ++j) {
        int koff = ((j < 4) ? 0 : 16) + g * 8 + 2 * (j & 3);
        float x0 = Vn[(size_t)(k0 + koff) * D_DIM + d];
        float x1 = Vn[(size_t)(k0 + koff + 1) * D_DIM + d];
        split_pack(x0, x1, bh.u[j], bl.u[j]);
    }
}

__device__ __forceinline__ v8f wmma_bf16(const Frag& a, const Frag& b, v8f c) {
    return __builtin_amdgcn_wmma_f32_16x16x32_bf16(
        /*neg_a=*/false, a.v, /*neg_b=*/false, b.v,
        /*c_mod=*/(short)0, c, /*reuse_a=*/false, /*reuse_b=*/false);
}

// hi*hi + hi*lo + lo*hi (drop lo*lo, ~2^-16 relative) — fp32-accurate product
__device__ __forceinline__ v8f wmma_split(const Frag& ah, const Frag& al,
                                          const Frag& bh, const Frag& bl, v8f c) {
    c = wmma_bf16(ah, bh, c);
    c = wmma_bf16(ah, bl, c);
    c = wmma_bf16(al, bh, c);
    return c;
}

// ---- LDS arena layout (130.25 KB, aliased across phases) --------------------
// [0      , 65536 ) : slot f32[32][512]  -> first 32 KB reused as pv f32[32][256]
// [65536  , 131072) : apacked_hi/lo ushort[32][256] (GEMM1)
//                     -> opacked_hi/lo ushort[32][512] (GEMM2)
// [131072 , 133376) : qs[256], vec[256], betal[32], beta_s[32]
#define SMEM_BYTES (65536 + 65536 + 2048 + 256)

__global__ __launch_bounds__(256)
void qpr_kernel(const float* __restrict__ q,  const float* __restrict__ K,
                const float* __restrict__ V,  const float* __restrict__ pq,
                const float* __restrict__ Wq, const float* __restrict__ Wc,
                const float* __restrict__ bc, const float* __restrict__ Wr,
                const float* __restrict__ br,
                float* __restrict__ out_logits, float* __restrict__ out_recon,
                float* __restrict__ out_eff) {
    __shared__ __align__(16) char smem[SMEM_BYTES];
    float* slot           = (float*)smem;                        // [32][512]
    float* pv             = (float*)smem;                        // [32][256] (alias)
    unsigned short* aph   = (unsigned short*)(smem + 65536);     // [32][256]
    unsigned short* apl   = aph + P_POOL * D_DIM;
    unsigned short* oph   = (unsigned short*)(smem + 65536);     // [32][512] (alias)
    unsigned short* opl   = oph + P_POOL * S_KV;
    float* qs             = (float*)(smem + 131072);             // [256]
    float* vec            = qs + D_DIM;                          // [256]
    float* betal          = vec + D_DIM;                         // [32]
    float* beta_s         = betal + P_POOL;                      // [32]

    const int n    = blockIdx.x;
    const int t    = threadIdx.x;
    const int lane = t & 31;
    const int wave = t >> 5;
    const int g    = (lane >> 4) & 1;
    const int col  = lane & 15;
    const float scale = 0.0625f; // 1/sqrt(256)

    const float* Kn = K + (size_t)n * S_KV * D_DIM;
    const float* Vn = V + (size_t)n * S_KV * D_DIM;

    // ---- q_proj = Wq @ q[n] -------------------------------------------------
    qs[t] = q[(size_t)n * D_DIM + t];
    __syncthreads();
    {
        const float* wr = Wq + (size_t)t * D_DIM;
        float acc = 0.f;
        for (int d = 0; d < D_DIM; ++d) acc += wr[d] * qs[d];
        vec[t] = acc;
    }
    __syncthreads();

    // ---- beta logits; pool_q built DIRECTLY as packed bf16 hi/lo ------------
    if (t < P_POOL) {
        const float* pr = pq + (size_t)t * D_DIM;
        float acc = 0.f;
        for (int d = 0; d < D_DIM; ++d) acc += pr[d] * vec[d];
        betal[t] = acc * scale;
    }
    for (int idx = t; idx < P_POOL * D_DIM; idx += 256) {
        int p = idx >> 8, d = idx & (D_DIM - 1);
        float x = (vec[d] + pq[(size_t)p * D_DIM + d]) * scale;
        split1(x, aph[idx], apl[idx]);
    }
    __syncthreads();

    if (t < P_POOL) {
        float m = -1e30f;
        for (int p = 0; p < P_POOL; ++p) m = fmaxf(m, betal[p]);
        float ssum = 0.f;
        for (int p = 0; p < P_POOL; ++p) ssum += __expf(betal[p] - m);
        beta_s[t] = __expf(betal[t] - m) / ssum;
    }

    // ---- GEMM1: slot_logits = pool_q(32x256) x Kn^T ------------------------
    // each wave owns a 16-wide s-tile; both 16-row m-tiles share the B fragment
    for (int pass = 0; pass < 4; ++pass) {
        int s0 = pass * 128 + wave * 16;
        v8f c0 = {}; v8f c1 = {};
        for (int k0 = 0; k0 < D_DIM; k0 += 32) {
            Frag bh, bl, ah, al;
            load_bK(Kn, s0 + col, k0, g, bh, bl);
            load_a_packed(aph, col, D_DIM, k0, g, ah);
            load_a_packed(apl, col, D_DIM, k0, g, al);
            c0 = wmma_split(ah, al, bh, bl, c0);
            load_a_packed(aph, 16 + col, D_DIM, k0, g, ah);
            load_a_packed(apl, 16 + col, D_DIM, k0, g, al);
            c1 = wmma_split(ah, al, bh, bl, c1);
        }
#pragma unroll
        for (int j = 0; j < 8; ++j) {   // D layout: row = 8g + j, col = lane&15
            slot[(g * 8 + j) * S_KV + s0 + col]      = c0[j];
            slot[(16 + g * 8 + j) * S_KV + s0 + col] = c1[j];
        }
    }
    __syncthreads();

    // ---- softmax over S per row (omega, normalized in place) ---------------
    for (int i = 0; i < 4; ++i) {
        int p = wave * 4 + i;
        float m = -1e30f;
        for (int s = lane; s < S_KV; s += 32) m = fmaxf(m, slot[p * S_KV + s]);
        for (int off = 16; off; off >>= 1) m = fmaxf(m, __shfl_xor(m, off, 32));
        float ssum = 0.f;
        for (int s = lane; s < S_KV; s += 32) {
            float e = __expf(slot[p * S_KV + s] - m);
            slot[p * S_KV + s] = e;
            ssum += e;
        }
        for (int off = 16; off; off >>= 1) ssum += __shfl_xor(ssum, off, 32);
        float inv = 1.f / ssum;
        for (int s = lane; s < S_KV; s += 32) slot[p * S_KV + s] *= inv;
    }
    __syncthreads();

    // ---- effective_weights + pack omega to bf16 hi/lo (apacked is dead) ----
    for (int s = t; s < S_KV; s += 256) {
        float acc = 0.f;
        for (int p = 0; p < P_POOL; ++p) acc += beta_s[p] * slot[p * S_KV + s];
        out_eff[(size_t)n * S_KV + s] = acc;
    }
    for (int idx = t; idx < P_POOL * S_KV; idx += 256) {
        split1(slot[idx], oph[idx], opl[idx]);
    }
    __syncthreads();   // slot is dead from here; its low 32 KB becomes pv

    // ---- GEMM2: pool_values = omega(32x512) x Vn(512x256) ------------------
    for (int cti = 0; cti < 2; ++cti) {
        int d0 = (wave + cti * 8) * 16;
        v8f c0 = {}; v8f c1 = {};
        for (int k0 = 0; k0 < S_KV; k0 += 32) {
            Frag bh, bl, ah0, al0, ah1, al1;
            load_bV(Vn, d0 + col, k0, g, bh, bl);
            load_a_packed(oph, col,      S_KV, k0, g, ah0);
            load_a_packed(opl, col,      S_KV, k0, g, al0);
            load_a_packed(oph, 16 + col, S_KV, k0, g, ah1);
            load_a_packed(opl, 16 + col, S_KV, k0, g, al1);
            c0 = wmma_split(ah0, al0, bh, bl, c0);
            c1 = wmma_split(ah1, al1, bh, bl, c1);
        }
#pragma unroll
        for (int j = 0; j < 8; ++j) {
            pv[(g * 8 + j) * D_DIM + d0 + col]      = c0[j];
            pv[(16 + g * 8 + j) * D_DIM + d0 + col] = c1[j];
        }
    }
    __syncthreads();

    // ---- summary[d] = sum_p beta[p] * pv[p][d] -----------------------------
    {
        float acc = 0.f;
        for (int p = 0; p < P_POOL; ++p) acc += beta_s[p] * pv[p * D_DIM + t];
        vec[t] = acc;   // q_proj is dead; reuse for summary
    }
    __syncthreads();

    // ---- recon = Wr @ summary + br -----------------------------------------
    {
        const float* wr = Wr + (size_t)t * D_DIM;
        float acc = br[t];
        for (int d = 0; d < D_DIM; ++d) acc += wr[d] * vec[d];
        out_recon[(size_t)n * D_DIM + t] = acc;
    }
    // ---- logits = Wc @ summary + bc ----------------------------------------
    for (int j = t; j < C_CLS; j += 256) {
        const float* wcr = Wc + (size_t)j * D_DIM;
        float acc = bc[j];
        for (int d = 0; d < D_DIM; ++d) acc += wcr[d] * vec[d];
        out_logits[(size_t)n * C_CLS + j] = acc;
    }
}

extern "C" void kernel_launch(void* const* d_in, const int* in_sizes, int n_in,
                              void* d_out, int out_size, void* d_ws, size_t ws_size,
                              hipStream_t stream) {
    const float* q  = (const float*)d_in[0];
    const float* K  = (const float*)d_in[1];
    const float* V  = (const float*)d_in[2];
    const float* pq = (const float*)d_in[3];
    const float* Wq = (const float*)d_in[4];
    const float* Wc = (const float*)d_in[5];
    const float* bc = (const float*)d_in[6];
    const float* Wr = (const float*)d_in[7];
    const float* br = (const float*)d_in[8];

    float* out        = (float*)d_out;
    float* out_logits = out;                                             // [N, C]
    float* out_recon  = out + (size_t)N_Q * C_CLS;                       // [N, D]
    float* out_eff    = out + (size_t)N_Q * C_CLS + (size_t)N_Q * D_DIM; // [N, S]

    hipLaunchKernelGGL(qpr_kernel, dim3(N_Q), dim3(256), 0, stream,
                       q, K, V, pq, Wq, Wc, bc, Wr, br,
                       out_logits, out_recon, out_eff);
}
